// NoTPAttention_11742440587309
// MI455X (gfx1250) — matrix operand order repaired and checked by reading, not compiled, not run on hardware
//
#include <hip/hip_runtime.h>
#include <hip/hip_bf16.h>

// ---------------------------------------------------------------------------
// Types for CDNA5 WMMA (gfx1250, wave32)
// ---------------------------------------------------------------------------
typedef __attribute__((ext_vector_type(16))) __bf16 v16bf;
typedef __attribute__((ext_vector_type(8)))  float  v8f;

typedef unsigned short bf16_t;   // raw bf16 storage (bit pattern only)

union FragAB {
    uint4 u[2];   // 32 bytes = 16 bf16 values per lane
    v16bf v;
};

__device__ __forceinline__ v8f wmma_bf16(const FragAB& a, const FragAB& b, v8f c) {
    // D(16x16 f32) = A(16x32 bf16) * B(32x16 bf16) + C
    return __builtin_amdgcn_wmma_f32_16x16x32_bf16(
        /*neg_a=*/false, a.v, /*neg_b=*/false, b.v,
        /*c_mod=*/(short)0, c, /*reuse_a=*/false, /*reuse_b=*/false);
}

__device__ __forceinline__ v8f zero8() {
    v8f z = {0.f, 0.f, 0.f, 0.f, 0.f, 0.f, 0.f, 0.f};
    return z;
}

// float -> bf16 (round-to-nearest-even), pure integer manipulation
__device__ __forceinline__ bf16_t f2bf(float f) {
    union { float f; unsigned u; } x;
    x.f = f;
    unsigned r = x.u + 0x7FFFu + ((x.u >> 16) & 1u);
    return (bf16_t)(r >> 16);
}

// ---------------------------------------------------------------------------
// Tensor Data Mover support (gfx1250). Guarded so the file still compiles
// (and the host pass parses) when the builtin is unavailable.
// ---------------------------------------------------------------------------
#if defined(__has_builtin)
#  if __has_builtin(__builtin_amdgcn_tensor_load_to_lds)
#    define HAVE_TDM 1
#  endif
#endif
#ifndef HAVE_TDM
#  define HAVE_TDM 0
#endif

#if __has_include(<hip/amd_detail/amd_gfx1250_TDM.h>)
#  define TDM_SIX_ARGS 1   // therock-10.0 headers -> 6-arg builtin
#else
#  define TDM_SIX_ARGS 0   // ROCm 7.2 -> 5-arg builtin
#endif

#if HAVE_TDM
typedef __attribute__((ext_vector_type(4))) unsigned tdm_u32x4;
typedef __attribute__((ext_vector_type(8))) int      tdm_i32x8;
typedef __attribute__((ext_vector_type(4))) int      tdm_i32x4;
typedef __attribute__((address_space(3))) unsigned char lds_byte_t;

__device__ __forceinline__ unsigned lds_off_of(const void* p) {
    // AS3 pointers are 32-bit LDS byte offsets on AMDGPU
    return (unsigned)(unsigned long long)(lds_byte_t*)p;
}

__device__ __forceinline__ void tdm_wait0() {
#  if __has_builtin(__builtin_amdgcn_s_wait_tensorcnt)
    __builtin_amdgcn_s_wait_tensorcnt(0);
#  else
    asm volatile("s_wait_tensorcnt 0x0" ::: "memory");
#  endif
}

// 2-D tile DMA: global (row-major, row_stride elems of 2B) -> LDS, with LDS
// row padding expressed via the D# pad fields.
//   pad_interval_code: 3 -> every 16 DWORDs, 5 -> every 64 DWORDs
//   pad_amount_code:   3 -> insert 4 DWORDs (16B)
__device__ __forceinline__ void tdm_load_tile_2d(
    unsigned lds_byte_off, const void* global_tile,
    unsigned tile_d0_elems, unsigned tile_d1_rows,
    unsigned long long row_stride_elems,
    unsigned pad_interval_code, unsigned pad_amount_code)
{
    unsigned long long ga = (unsigned long long)(uintptr_t)global_tile;

    // D# group 0: count=1, lds_addr, global_addr[56:0], type=2 ("image")
    tdm_u32x4 g0;
    g0[0] = 1u;
    g0[1] = lds_byte_off;
    g0[2] = (unsigned)(ga & 0xFFFFFFFFull);
    g0[3] = (unsigned)((ga >> 32) & 0x01FFFFFFull) | (2u << 30);

    // D# group 1: data_size=1 (2B), pad_enable, pad codes,
    // tensor dims == tile dims (no OOB), dim0 stride = row stride
    tdm_i32x8 g1;
    g1[0] = (int)((1u << 16) | (1u << 20) |
                  (pad_interval_code << 22) | (pad_amount_code << 25));
    g1[1] = (int)((tile_d0_elems & 0xFFFFu) << 16);                    // tensor_dim0[15:0]
    g1[2] = (int)(((tile_d0_elems >> 16) & 0xFFFFu) |
                  ((tile_d1_rows & 0xFFFFu) << 16));                   // dim0[31:16] | dim1[15:0]
    g1[3] = (int)(((tile_d1_rows >> 16) & 0xFFFFu) |
                  ((tile_d0_elems & 0xFFFFu) << 16));                  // dim1[31:16] | tile_dim0
    g1[4] = (int)(tile_d1_rows & 0xFFFFu);                             // tile_dim1 (tile_dim2=0)
    g1[5] = (int)(unsigned)(row_stride_elems & 0xFFFFFFFFull);         // dim0_stride[31:0]
    g1[6] = (int)(unsigned)((row_stride_elems >> 32) & 0xFFFFull);     // dim0_stride[47:32]
    g1[7] = 0;

    tdm_i32x4 z4 = {0, 0, 0, 0};
#  if TDM_SIX_ARGS
    tdm_i32x8 z8 = {0, 0, 0, 0, 0, 0, 0, 0};
    __builtin_amdgcn_tensor_load_to_lds(g0, g1, z4, z4, z8, 0);
#  else
    __builtin_amdgcn_tensor_load_to_lds(g0, g1, z4, z4, 0);
#  endif
}
#endif  // HAVE_TDM

// ---------------------------------------------------------------------------
// f32 -> bf16 conversion (vectorized x4)
// ---------------------------------------------------------------------------
__global__ __launch_bounds__(256)
void cvt_f32_to_bf16_kernel(const float4* __restrict__ in, uint2* __restrict__ out, int n4) {
    int i = blockIdx.x * blockDim.x + threadIdx.x;
    if (i < n4) {
        float4 f = in[i];
        unsigned h0 = f2bf(f.x), h1 = f2bf(f.y), h2 = f2bf(f.z), h3 = f2bf(f.w);
        uint2 o;
        o.x = h0 | (h1 << 16);
        o.y = h2 | (h3 << 16);
        out[i] = o;
    }
}

// ---------------------------------------------------------------------------
// Tiled NT GEMM:  C[M,N] = A[M,K](bf16) * B[N,K](bf16)^T + bias[N]
// Block tile 128x128, BK=32, 256 threads = 8 waves, each wave 32x64 (2x4 WMMA)
// Double-buffered LDS; tiles staged by the Tensor Data Mover when available
// (DMA for tile i+1 overlaps the WMMAs of tile i; one barrier per K-step).
// ---------------------------------------------------------------------------
template <bool OUT_BF16>
__global__ __launch_bounds__(256)
void gemm_nt_bf16_kernel(const bf16_t* __restrict__ A,
                         const bf16_t* __restrict__ B,
                         const float*  __restrict__ bias,
                         void* __restrict__ Cout,
                         int M, int N, int K) {
    __shared__ __align__(16) bf16_t sA[2][128][40];  // 64B row + 16B pad
    __shared__ __align__(16) bf16_t sB[2][128][40];

    const int tid    = threadIdx.x;
    const int lane   = tid & 31;
    const int wave   = tid >> 5;
    const int lanelo = lane & 15;
    const int lanehi = lane >> 4;

    const int m0 = blockIdx.y * 128;
    const int n0 = blockIdx.x * 128;
    const int wm = (wave & 3) * 32;   // 4 waves along M
    const int wn = (wave >> 2) * 64;  // 2 waves along N

    v8f acc[2][4];
#pragma unroll
    for (int mi = 0; mi < 2; ++mi)
#pragma unroll
        for (int ni = 0; ni < 4; ++ni) acc[mi][ni] = zero8();

#if HAVE_TDM
    const unsigned sA_off[2] = { lds_off_of(&sA[0][0][0]), lds_off_of(&sA[1][0][0]) };
    const unsigned sB_off[2] = { lds_off_of(&sB[0][0][0]), lds_off_of(&sB[1][0][0]) };
    auto issue_tile = [&](int buf, int k0) {
        tdm_load_tile_2d(sA_off[buf], A + (size_t)m0 * K + k0, 32, 128,
                         (unsigned long long)K, 3, 3);
        tdm_load_tile_2d(sB_off[buf], B + (size_t)n0 * K + k0, 32, 128,
                         (unsigned long long)K, 3, 3);
    };
#else
    auto stage_tile = [&](int buf, int k0) {
#pragma unroll
        for (int i = 0; i < 2; ++i) {
            int idx = tid * 2 + i;        // 0..511
            int r = idx >> 2;             // 0..127
            int c = idx & 3;              // chunk
            const uint4* ga = (const uint4*)(A + (size_t)(m0 + r) * K + k0);
            ((uint4*)&sA[buf][r][0])[c] = ga[c];
            const uint4* gb = (const uint4*)(B + (size_t)(n0 + r) * K + k0);
            ((uint4*)&sB[buf][r][0])[c] = gb[c];
        }
    };
#endif

    // prologue: bring in tile 0
#if HAVE_TDM
    if (wave == 0) issue_tile(0, 0);
#else
    stage_tile(0, 0);
#endif

    const int nk = K >> 5;
    for (int it = 0; it < nk; ++it) {
        const int buf = it & 1;
        const int k0n = (it + 1) << 5;

#if HAVE_TDM
        if (wave == 0) tdm_wait0();     // tile it fully in LDS
#endif
        __syncthreads();                // ...and buf^1 readers retired

        // fragments of tile it
        FragAB afrag[2], bfrag[4];
#pragma unroll
        for (int mi = 0; mi < 2; ++mi) {
            const uint4* p = (const uint4*)&sA[buf][wm + mi * 16 + lanelo][0];
            afrag[mi].u[0] = p[lanehi];       // K = lanehi*8 .. +7
            afrag[mi].u[1] = p[lanehi + 2];   // K = 16 + lanehi*8 .. +7
        }
#pragma unroll
        for (int ni = 0; ni < 4; ++ni) {
            const uint4* p = (const uint4*)&sB[buf][wn + ni * 16 + lanelo][0];
            bfrag[ni].u[0] = p[lanehi];       // K = lanehi*8 .. +7
            bfrag[ni].u[1] = p[lanehi + 2];   // K = 16 + lanehi*8 .. +7
        }

        // kick off tile it+1 (overlaps the WMMAs below)
        if (it + 1 < nk) {
#if HAVE_TDM
            if (wave == 0) issue_tile(buf ^ 1, k0n);
#else
            stage_tile(buf ^ 1, k0n);
#endif
        }

#pragma unroll
        for (int mi = 0; mi < 2; ++mi)
#pragma unroll
            for (int ni = 0; ni < 4; ++ni)
                acc[mi][ni] = wmma_bf16(afrag[mi], bfrag[ni], acc[mi][ni]);
    }

    // epilogue: C/D layout — VGPR j, lanes 0-15 -> row j, lanes 16-31 -> row j+8
#pragma unroll
    for (int mi = 0; mi < 2; ++mi) {
#pragma unroll
        for (int ni = 0; ni < 4; ++ni) {
            int n = n0 + wn + ni * 16 + lanelo;
            float bv = bias[n];
#pragma unroll
            for (int j = 0; j < 8; ++j) {
                int m = m0 + wm + mi * 16 + j + 8 * lanehi;
                float val = acc[mi][ni][j] + bv;
                if (OUT_BF16) {
                    ((bf16_t*)Cout)[(size_t)m * N + n] = f2bf(val);
                } else {
                    ((float*)Cout)[(size_t)m * N + n] = val;
                }
            }
        }
    }
}

// ---------------------------------------------------------------------------
// Flash attention forward (bf16 in/out, fp32 accumulate / softmax)
// grid: (S/128, B*NH); block 256 = 8 waves; each wave = 16 q-rows
// xqkv layout: [B][S][3][NH][HD] bf16, attn_out: [B][S][NH*HD] bf16
// Double-buffered K (TDM strided 2-D tile) + V (manual transpose) staging;
// next tile's DMA/loads overlap the current tile's WMMAs and softmax.
// ---------------------------------------------------------------------------
__global__ __launch_bounds__(256)
void attn_fwd_kernel(const bf16_t* __restrict__ xqkv, bf16_t* __restrict__ attn_out) {
    constexpr int S = 2048, H = 2048, HD = 128;
    constexpr int ROWSTRIDE = 3 * H;  // bf16 elems between consecutive s

    __shared__ __align__(16) bf16_t sK[2][32][136];   // 256B row + 16B pad
    __shared__ __align__(16) bf16_t sVt[2][128][40];  // transposed V: [d][kv]
    __shared__ __align__(16) bf16_t sP[8][16][40];    // per-wave P tile 16x32

    const int tid    = threadIdx.x;
    const int lane   = tid & 31;
    const int wave   = tid >> 5;
    const int lanelo = lane & 15;
    const int lanehi = lane >> 4;

    const int bh = blockIdx.y;
    const int b  = bh >> 4;
    const int h  = bh & 15;
    const int q0 = blockIdx.x * 128 + wave * 16;

    const size_t base = (size_t)b * S * ROWSTRIDE + (size_t)h * HD;
    const bf16_t* Qp = xqkv + base;           // part 0
    const bf16_t* Kp = xqkv + base + H;       // part 1
    const bf16_t* Vp = xqkv + base + 2 * H;   // part 2

    // Q fragments resident in registers: A-matrix 16x32 per K-slice, 4 slices
    FragAB qf[4];
    {
        const uint4* qr = (const uint4*)(Qp + (size_t)(q0 + lanelo) * ROWSTRIDE);
#pragma unroll
        for (int f = 0; f < 4; ++f) {
            qf[f].u[0] = qr[f * 4 + lanehi];
            qf[f].u[1] = qr[f * 4 + lanehi + 2];
        }
    }

    v8f   o[8];
    float mrow[8], lrow[8];
#pragma unroll
    for (int g = 0; g < 8; ++g) o[g] = zero8();
#pragma unroll
    for (int j = 0; j < 8; ++j) { mrow[j] = -1.0e30f; lrow[j] = 0.f; }

    const float scale = 0.0883883476483184f;  // 1/sqrt(128)

#if HAVE_TDM
    const unsigned sK_off[2] = { lds_off_of(&sK[0][0][0]), lds_off_of(&sK[1][0][0]) };
#endif

    // stage V tile transposed (and K tile when no TDM) into buffer `buf`
    auto stage_v = [&](int buf, int kv0) {
#pragma unroll
        for (int i = 0; i < 2; ++i) {
            int idx = tid * 2 + i;   // 0..511
            int r = idx >> 4;        // kv row 0..31
            int c = idx & 15;        // 8-elem chunk along d
#if !HAVE_TDM
            uint4 kk = *(const uint4*)(Kp + (size_t)(kv0 + r) * ROWSTRIDE + c * 8);
            ((uint4*)&sK[buf][r][0])[c] = kk;
#endif
            union { uint4 u; bf16_t e[8]; } vv;
            vv.u = *(const uint4*)(Vp + (size_t)(kv0 + r) * ROWSTRIDE + c * 8);
#pragma unroll
            for (int e = 0; e < 8; ++e) sVt[buf][c * 8 + e][r] = vv.e[e];
        }
    };

    // prologue: bring in tile 0
#if HAVE_TDM
    if (wave == 0)
        tdm_load_tile_2d(sK_off[0], Kp, 128, 32, (unsigned long long)ROWSTRIDE, 5, 3);
#endif
    stage_v(0, 0);

    constexpr int NT = S / 32;  // 64 kv tiles
    for (int it = 0; it < NT; ++it) {
        const int buf = it & 1;
        const int kv0n = (it + 1) * 32;

#if HAVE_TDM
        if (wave == 0) tdm_wait0();     // K tile it fully in LDS
#endif
        __syncthreads();                // ...and buf^1 readers retired

        // kick off tile it+1 (overlaps WMMAs + softmax below)
        if (it + 1 < NT) {
#if HAVE_TDM
            if (wave == 0)
                tdm_load_tile_2d(sK_off[buf ^ 1], Kp + (size_t)kv0n * ROWSTRIDE,
                                 128, 32, (unsigned long long)ROWSTRIDE, 5, 3);
#endif
            stage_v(buf ^ 1, kv0n);
        }

        // ---- S = Q @ K^T : two 16x16 tiles (kv cols 0-15, 16-31) ----
        v8f sf[2];
#pragma unroll
        for (int nt = 0; nt < 2; ++nt) {
            sf[nt] = zero8();
            const uint4* p = (const uint4*)&sK[buf][nt * 16 + lanelo][0];
#pragma unroll
            for (int f = 0; f < 4; ++f) {
                FragAB kb;
                kb.u[0] = p[f * 4 + lanehi];
                kb.u[1] = p[f * 4 + lanehi + 2];
                sf[nt] = wmma_bf16(qf[f], kb, sf[nt]);
            }
        }

        // ---- online softmax (rows spread across 16-lane groups) ----
        float p0[8], p1[8];
#pragma unroll
        for (int j = 0; j < 8; ++j) {
            float s0 = sf[0][j] * scale;
            float s1 = sf[1][j] * scale;
            float t = fmaxf(s0, s1);
#pragma unroll
            for (int off = 1; off < 16; off <<= 1)
                t = fmaxf(t, __shfl_xor(t, off, 16));
            float mn = fmaxf(mrow[j], t);
            float e0 = __expf(s0 - mn);
            float e1 = __expf(s1 - mn);
            float rs = e0 + e1;
#pragma unroll
            for (int off = 1; off < 16; off <<= 1)
                rs += __shfl_xor(rs, off, 16);
            float r = __expf(mrow[j] - mn);
            lrow[j] = lrow[j] * r + rs;
            mrow[j] = mn;
#pragma unroll
            for (int g = 0; g < 8; ++g) o[g][j] *= r;
            p0[j] = e0;
            p1[j] = e1;
        }

        // ---- C-layout -> A-layout for P via wave-private LDS ----
#pragma unroll
        for (int j = 0; j < 8; ++j) {
            sP[wave][j + 8 * lanehi][lanelo]      = f2bf(p0[j]);
            sP[wave][j + 8 * lanehi][16 + lanelo] = f2bf(p1[j]);
        }
        FragAB pf;
        {
            const uint4* p = (const uint4*)&sP[wave][lanelo][0];
            pf.u[0] = p[lanehi];
            pf.u[1] = p[lanehi + 2];
        }

        // ---- O += P @ V : 8 n-tiles over d ----
#pragma unroll
        for (int g = 0; g < 8; ++g) {
            FragAB vb;
            const uint4* p = (const uint4*)&sVt[buf][g * 16 + lanelo][0];
            vb.u[0] = p[lanehi];
            vb.u[1] = p[lanehi + 2];
            o[g] = wmma_bf16(pf, vb, o[g]);
        }
    }

    // ---- normalize and write attn output (bf16) ----
#pragma unroll
    for (int g = 0; g < 8; ++g) {
#pragma unroll
        for (int j = 0; j < 8; ++j) {
            int srow = q0 + j + 8 * lanehi;
            int col  = h * HD + g * 16 + lanelo;
            attn_out[((size_t)b * S + srow) * H + col] = f2bf(o[g][j] / lrow[j]);
        }
    }
}

// ---------------------------------------------------------------------------
// Launch
// ---------------------------------------------------------------------------
extern "C" void kernel_launch(void* const* d_in, const int* in_sizes, int n_in,
                              void* d_out, int out_size, void* d_ws, size_t ws_size,
                              hipStream_t stream) {
    const float* x     = (const float*)d_in[0];   // (2,2048,2048)
    const float* w_qkv = (const float*)d_in[1];   // (6144,2048)
    const float* b_qkv = (const float*)d_in[2];   // (6144,)
    const float* w_out = (const float*)d_in[3];   // (2048,2048)
    const float* b_out = (const float*)d_in[4];   // (2048,)
    float* out = (float*)d_out;                   // (2,2048,2048) f32

    constexpr int B = 2, S = 2048, Hd = 2048, NH = 16;
    constexpr int M = B * S;          // 4096
    constexpr int N_qkv = 3 * Hd;     // 6144

    constexpr size_t nx   = (size_t)M * Hd;
    constexpr size_t nwq  = (size_t)N_qkv * Hd;
    constexpr size_t nwo  = (size_t)Hd * Hd;
    constexpr size_t nqkv = (size_t)M * N_qkv;

    char* ws = (char*)d_ws;
    bf16_t* x_bf   = (bf16_t*)ws;  ws += nx * 2;
    bf16_t* wq_bf  = (bf16_t*)ws;  ws += nwq * 2;
    bf16_t* wo_bf  = (bf16_t*)ws;  ws += nwo * 2;
    bf16_t* qkv_bf = (bf16_t*)ws;  ws += nqkv * 2;
    bf16_t* ao_bf  = (bf16_t*)ws;  ws += nx * 2;

    {
        int n4 = (int)(nx / 4);
        cvt_f32_to_bf16_kernel<<<(n4 + 255) / 256, 256, 0, stream>>>(
            (const float4*)x, (uint2*)x_bf, n4);
    }
    {
        int n4 = (int)(nwq / 4);
        cvt_f32_to_bf16_kernel<<<(n4 + 255) / 256, 256, 0, stream>>>(
            (const float4*)w_qkv, (uint2*)wq_bf, n4);
    }
    {
        int n4 = (int)(nwo / 4);
        cvt_f32_to_bf16_kernel<<<(n4 + 255) / 256, 256, 0, stream>>>(
            (const float4*)w_out, (uint2*)wo_bf, n4);
    }

    // QKV projection: xqkv = x @ w_qkv^T + b_qkv  (output bf16)
    gemm_nt_bf16_kernel<true><<<dim3(N_qkv / 128, M / 128), 256, 0, stream>>>(
        x_bf, wq_bf, b_qkv, (void*)qkv_bf, M, N_qkv, Hd);

    // attention
    attn_fwd_kernel<<<dim3(S / 128, B * NH), 256, 0, stream>>>(qkv_bf, ao_bf);

    // output projection: out = attn @ w_out^T + b_out (output f32)
    gemm_nt_bf16_kernel<false><<<dim3(Hd / 128, M / 128), 256, 0, stream>>>(
        ao_bf, wo_bf, b_out, (void*)out, M, Hd, Hd);
}
